// Attention_68006512165491
// MI455X (gfx1250) — matrix operand order repaired and checked by reading, not compiled, not run on hardware
//
#include <hip/hip_runtime.h>
#include <hip/hip_bf16.h>

typedef __attribute__((ext_vector_type(16))) _Float16 v16h;
typedef __attribute__((ext_vector_type(8)))  _Float16 v8h;
typedef __attribute__((ext_vector_type(4)))  _Float16 v4h;
typedef __attribute__((ext_vector_type(8)))  float    v8f;

#define SHUF16(lo, hi) __builtin_shufflevector((lo), (hi), 0,1,2,3,4,5,6,7,8,9,10,11,12,13,14,15)
#define WMMA_F16(a, b, c) \
  __builtin_amdgcn_wmma_f32_16x16x32_f16(false, (a), false, (b), (short)0, (c), false, false)

namespace {
constexpr int kB  = 4;
constexpr int kN  = 1024;
constexpr int kC  = 768;
constexpr int kH  = 12;
constexpr int kHD = 64;
constexpr int kM  = kB * kN;   // 4096 rows for x / out
constexpr int kC3 = 3 * kC;    // 2304
constexpr int SCW = 1028;      // padded LDS score row stride (floats) to avoid bank conflicts
}

// ---------------------------------------------------------------------------
// f32 -> f16 conversion (vectorized x4)
// ---------------------------------------------------------------------------
__global__ void cvt_f32_to_f16(const float* __restrict__ in, _Float16* __restrict__ out, int n4) {
  int i = blockIdx.x * blockDim.x + threadIdx.x;
  if (i >= n4) return;
  float4 f = ((const float4*)in)[i];
  v4h h = { (_Float16)f.x, (_Float16)f.y, (_Float16)f.z, (_Float16)f.w };
  ((v4h*)out)[i] = h;
}

// ---------------------------------------------------------------------------
// QKV GEMM: y = x @ qkv_w^T  (M=4096, N=2304, K=768), f16 in / f32 acc.
// Register-blocked: each wave computes a 16x64 strip (A fragment reused 4x).
// Scatters results into q[b,h,n,d], k[b,h,n,d], vT[b,h,d,n] (all f16).
// ---------------------------------------------------------------------------
__global__ __launch_bounds__(128) void qkv_gemm_kernel(
    const _Float16* __restrict__ xh, const _Float16* __restrict__ wh,
    _Float16* __restrict__ qh, _Float16* __restrict__ kh, _Float16* __restrict__ vTh) {
  const int lane = threadIdx.x & 31;
  const int tile = blockIdx.x * 4 + (threadIdx.x >> 5);  // 256*36 = 9216 strips
  const int tm   = tile / (kC3 / 64);
  const int tn   = tile % (kC3 / 64);
  const int row0 = tm * 16, col0 = tn * 64;
  const int l16  = lane & 15, lhi = lane >> 4;

  const _Float16* ap = xh + (size_t)(row0 + l16) * kC + lhi * 16;  // A: rows of x
  const _Float16* bp = wh + (size_t)(col0 + l16) * kC + lhi * 16;  // B: rows of qkv_w
  v8f acc[4] = {};
  for (int kk = 0; kk < kC; kk += 32) {
    if (kk + 32 < kC) {
      __builtin_prefetch(ap + kk + 32, 0, 1);
      __builtin_prefetch(bp + kk + 32, 0, 1);
    }
    v8h alo = *(const v8h*)(ap + kk);
    v8h ahi = *(const v8h*)(ap + kk + 8);
    v16h a  = SHUF16(alo, ahi);
    v16h b0 = *(const v16h*)(bp + kk);
    v16h b1 = *(const v16h*)(bp + (size_t)16 * kC + kk);
    v16h b2 = *(const v16h*)(bp + (size_t)32 * kC + kk);
    v16h b3 = *(const v16h*)(bp + (size_t)48 * kC + kk);
    acc[0] = WMMA_F16(a, b0, acc[0]);
    acc[1] = WMMA_F16(a, b1, acc[1]);
    acc[2] = WMMA_F16(a, b2, acc[2]);
    acc[3] = WMMA_F16(a, b3, acc[3]);
  }

#pragma unroll
  for (int u = 0; u < 4; ++u) {
    const int ncol = col0 + u * 16 + l16;
    const int t    = ncol / kC;          // 0=q, 1=k, 2=v
    const int rem  = ncol - t * kC;
    const int h    = rem / kHD;
    const int d    = rem - h * kHD;
#pragma unroll
    for (int j = 0; j < 8; ++j) {
      const int m  = row0 + j + 8 * lhi;
      const int bi = m >> 10;
      const int ni = m & (kN - 1);
      const int bh = bi * kH + h;
      const _Float16 val = (_Float16)acc[u][j];
      if (t == 0)      qh[((size_t)bh * kN + ni) * kHD + d] = val;
      else if (t == 1) kh[((size_t)bh * kN + ni) * kHD + d] = val;
      else             vTh[((size_t)bh * kHD + d) * kN + ni] = val;   // store V transposed
    }
  }
}

// ---------------------------------------------------------------------------
// Attention: one 16-row tile per 128-thread block (4 waves).
//  phase 1: logits = q @ k^T * scale via WMMA (wave w -> cols [w*256, w*256+256))
//  phase1.5: per-row bias MLP (lane = hidden unit, shfl reduce)
//  phase 2: row softmax (exp kept in LDS, f32)
//  phase 3: single pass: write cur_attn, blend with prev_attn -> LDS
//  phase 4: out = blend @ V via WMMA (wave w -> d-tile w, full K)
// ---------------------------------------------------------------------------
__global__ __launch_bounds__(128) void attn_kernel(
    const _Float16* __restrict__ qh, const _Float16* __restrict__ kh,
    const _Float16* __restrict__ vTh, const float* __restrict__ prev,
    const float* __restrict__ w1, const float* __restrict__ b1,
    const float* __restrict__ w2, const float* __restrict__ b2,
    float* __restrict__ cur, _Float16* __restrict__ aoh) {
  __shared__ float s_scores[16 * SCW];
  __shared__ float s_bias[16][3];
  __shared__ float s_denom[16];

  const int bh   = blockIdx.x >> 6;   // (b*H + h)
  const int tile = blockIdx.x & 63;
  const int row0 = tile * 16;
  const int bidx = bh / kH;
  const int hidx = bh % kH;
  const int lane = threadIdx.x & 31;
  const int w    = threadIdx.x >> 5;
  const int l16  = lane & 15, lhi = lane >> 4;
  const float scale = 0.125f;  // hd^-0.5

  // ---- phase 1: logits --------------------------------------------------
  const _Float16* qp = qh + ((size_t)bh * kN + row0 + l16) * kHD + lhi * 16;
  v16h aq0, aq1;
  {
    v8h lo0 = *(const v8h*)(qp);
    v8h hi0 = *(const v8h*)(qp + 8);
    v8h lo1 = *(const v8h*)(qp + 32);
    v8h hi1 = *(const v8h*)(qp + 40);
    aq0 = SHUF16(lo0, hi0);
    aq1 = SHUF16(lo1, hi1);
  }
  for (int ct = 0; ct < 16; ++ct) {
    const int col0 = w * 256 + ct * 16;
    const _Float16* kp = kh + ((size_t)bh * kN + col0 + l16) * kHD + lhi * 16;
    v8f acc = {};
    v16h b0  = *(const v16h*)(kp);
    v16h b1f = *(const v16h*)(kp + 32);
    acc = WMMA_F16(aq0, b0, acc);
    acc = WMMA_F16(aq1, b1f, acc);
#pragma unroll
    for (int j = 0; j < 8; ++j)
      s_scores[(j + 8 * lhi) * SCW + col0 + l16] = acc[j] * scale;
  }

  // ---- phase 1.5: bias MLP for rows w*4 .. w*4+3 ------------------------
#pragma unroll
  for (int rr = 0; rr < 4; ++rr) {
    const int r = w * 4 + rr;
    const _Float16* qr = qh + ((size_t)bh * kN + row0 + r) * kHD;
    const _Float16* kr = kh + ((size_t)bh * kN + row0 + r) * kHD;
    const float* w1r = w1 + lane * (2 * kHD);  // hidden unit = lane (32 units)
    float s = b1[lane];
    for (int d2 = 0; d2 < kHD; ++d2) s += w1r[d2] * (float)qr[d2];
    for (int d2 = 0; d2 < kHD; ++d2) s += w1r[kHD + d2] * (float)kr[d2];
    s = fmaxf(s, 0.0f);
    float p0 = s * w2[lane];
    float p1 = s * w2[32 + lane];
    float p2 = s * w2[64 + lane];
    for (int off = 16; off; off >>= 1) {
      p0 += __shfl_xor(p0, off, 32);
      p1 += __shfl_xor(p1, off, 32);
      p2 += __shfl_xor(p2, off, 32);
    }
    if (lane == 0) {
      s_bias[r][0] = fabsf(p0 + b2[0]);
      s_bias[r][1] = fabsf(p1 + b2[1]);
      s_bias[r][2] = fabsf(p2 + b2[2]);
    }
  }
  __syncthreads();

  // ---- phase 2: softmax rows w*4 .. w*4+3 --------------------------------
  for (int rr = 0; rr < 4; ++rr) {
    const int r = w * 4 + rr;
    float mx = -3.0e38f;
    for (int c = lane; c < kN; c += 32) mx = fmaxf(mx, s_scores[r * SCW + c]);
    for (int off = 16; off; off >>= 1) mx = fmaxf(mx, __shfl_xor(mx, off, 32));
    float sum = 0.0f;
    for (int c = lane; c < kN; c += 32) {
      float e = __expf(s_scores[r * SCW + c] - mx);
      s_scores[r * SCW + c] = e;
      sum += e;
    }
    for (int off = 16; off; off >>= 1) sum += __shfl_xor(sum, off, 32);
    if (lane == 0) s_denom[r] = sum;
  }
  __syncthreads();

  // ---- phase 3: write cur_attn + blend with prev_attn --------------------
  const size_t pbase = ((size_t)bh * kN + row0) * kN;
  for (int r = 0; r < 16; ++r) {
    const float inv = 1.0f / s_denom[r];
    const float bb0 = 1.0f + s_bias[r][0];
    const float bb1 = s_bias[r][1];
    const float bb2 = s_bias[r][2] * (1.0f / kN);
#pragma unroll
    for (int cc = 0; cc < 2; ++cc) {
      const int c = w * 256 + cc * 128 + lane * 4;
      float4 e  = *(const float4*)&s_scores[r * SCW + c];
      float4 pv = *(const float4*)&prev[pbase + (size_t)r * kN + c];
      float4 cu, bl;
      cu.x = e.x * inv; cu.y = e.y * inv; cu.z = e.z * inv; cu.w = e.w * inv;
      bl.x = cu.x * bb0 + pv.x * bb1 - bb2;
      bl.y = cu.y * bb0 + pv.y * bb1 - bb2;
      bl.z = cu.z * bb0 + pv.z * bb1 - bb2;
      bl.w = cu.w * bb0 + pv.w * bb1 - bb2;
      *(float4*)&cur[pbase + (size_t)r * kN + c] = cu;
      *(float4*)&s_scores[r * SCW + c] = bl;
    }
  }
  __syncthreads();

  // ---- phase 4: out(16 x 64) = blend @ V; wave w owns d-tile w -----------
  v8f acc = {};
  const _Float16* vp = vTh + ((size_t)bh * kHD + w * 16 + l16) * kN + lhi * 16;
  for (int K0 = 0; K0 < kN; K0 += 32) {
    const float* sp = &s_scores[l16 * SCW + K0 + lhi * 16];
    float4 f0 = *(const float4*)(sp);
    float4 f1 = *(const float4*)(sp + 4);
    float4 f2 = *(const float4*)(sp + 8);
    float4 f3 = *(const float4*)(sp + 12);
    v16h a = { (_Float16)f0.x, (_Float16)f0.y, (_Float16)f0.z, (_Float16)f0.w,
               (_Float16)f1.x, (_Float16)f1.y, (_Float16)f1.z, (_Float16)f1.w,
               (_Float16)f2.x, (_Float16)f2.y, (_Float16)f2.z, (_Float16)f2.w,
               (_Float16)f3.x, (_Float16)f3.y, (_Float16)f3.z, (_Float16)f3.w };
    v16h b = *(const v16h*)(vp + K0);
    acc = WMMA_F16(a, b, acc);
  }
  // write head output into [B, N, C] f16 buffer for the proj GEMM
  _Float16* op = aoh + ((size_t)bidx * kN + row0) * kC + hidx * kHD + w * 16 + l16;
#pragma unroll
  for (int j = 0; j < 8; ++j)
    op[(size_t)(j + 8 * lhi) * kC] = (_Float16)acc[j];
}

// ---------------------------------------------------------------------------
// Output projection: out = attn_out @ proj_w^T + proj_b (M=4096, N=768, K=768)
// Register-blocked 16x64 strip per wave.
// ---------------------------------------------------------------------------
__global__ __launch_bounds__(128) void proj_gemm_kernel(
    const _Float16* __restrict__ ah, const _Float16* __restrict__ wh,
    const float* __restrict__ pb, float* __restrict__ out) {
  const int lane = threadIdx.x & 31;
  const int tile = blockIdx.x * 4 + (threadIdx.x >> 5);  // 256*12 = 3072 strips
  const int tm   = tile / (kC / 64);
  const int tn   = tile % (kC / 64);
  const int row0 = tm * 16, col0 = tn * 64;
  const int l16  = lane & 15, lhi = lane >> 4;

  const _Float16* ap = ah + (size_t)(row0 + l16) * kC + lhi * 16;
  const _Float16* bp = wh + (size_t)(col0 + l16) * kC + lhi * 16;
  v8f acc[4] = {};
  for (int kk = 0; kk < kC; kk += 32) {
    if (kk + 32 < kC) {
      __builtin_prefetch(ap + kk + 32, 0, 1);
      __builtin_prefetch(bp + kk + 32, 0, 1);
    }
    v8h alo = *(const v8h*)(ap + kk);
    v8h ahi = *(const v8h*)(ap + kk + 8);
    v16h a  = SHUF16(alo, ahi);
    v16h b0 = *(const v16h*)(bp + kk);
    v16h b1 = *(const v16h*)(bp + (size_t)16 * kC + kk);
    v16h b2 = *(const v16h*)(bp + (size_t)32 * kC + kk);
    v16h b3 = *(const v16h*)(bp + (size_t)48 * kC + kk);
    acc[0] = WMMA_F16(a, b0, acc[0]);
    acc[1] = WMMA_F16(a, b1, acc[1]);
    acc[2] = WMMA_F16(a, b2, acc[2]);
    acc[3] = WMMA_F16(a, b3, acc[3]);
  }
#pragma unroll
  for (int u = 0; u < 4; ++u) {
    const int ncol = col0 + u * 16 + l16;
    const float bias = pb[ncol];
#pragma unroll
    for (int j = 0; j < 8; ++j) {
      const int m = row0 + j + 8 * lhi;
      out[(size_t)m * kC + ncol] = acc[u][j] + bias;
    }
  }
}

// ---------------------------------------------------------------------------
extern "C" void kernel_launch(void* const* d_in, const int* in_sizes, int n_in,
                              void* d_out, int out_size, void* d_ws, size_t ws_size,
                              hipStream_t stream) {
  (void)in_sizes; (void)n_in; (void)out_size; (void)ws_size;
  const float* x      = (const float*)d_in[0];
  const float* prev   = (const float*)d_in[1];
  const float* qkv_w  = (const float*)d_in[2];
  const float* proj_w = (const float*)d_in[3];
  const float* proj_b = (const float*)d_in[4];
  const float* bp_w1  = (const float*)d_in[5];
  const float* bp_b1  = (const float*)d_in[6];
  const float* bp_w2  = (const float*)d_in[7];
  const float* bp_b2  = (const float*)d_in[8];

  float* out = (float*)d_out;                       // [4,1024,768]
  float* cur = out + (size_t)kM * kC;               // [4,12,1024,1024]

  // workspace layout (f16), all sub-offsets 32B aligned
  _Float16* xh  = (_Float16*)d_ws;                  // [4096,768]
  _Float16* wqh = xh  + (size_t)kM * kC;            // [2304,768]
  _Float16* wph = wqh + (size_t)kC3 * kC;           // [768,768]
  _Float16* qh  = wph + (size_t)kC * kC;            // [B*H,N,hd]
  _Float16* kh  = qh  + (size_t)kB * kH * kN * kHD;
  _Float16* vTh = kh  + (size_t)kB * kH * kN * kHD; // [B*H,hd,N]
  _Float16* aoh = vTh + (size_t)kB * kH * kN * kHD; // [4096,768]

  {
    int n4 = (kM * kC) / 4;
    cvt_f32_to_f16<<<(n4 + 255) / 256, 256, 0, stream>>>(x, xh, n4);
  }
  {
    int n4 = (kC3 * kC) / 4;
    cvt_f32_to_f16<<<(n4 + 255) / 256, 256, 0, stream>>>(qkv_w, wqh, n4);
  }
  {
    int n4 = (kC * kC) / 4;
    cvt_f32_to_f16<<<(n4 + 255) / 256, 256, 0, stream>>>(proj_w, wph, n4);
  }

  // QKV GEMM: (4096/16)*(2304/64) = 9216 strips, 4 waves/block
  qkv_gemm_kernel<<<9216 / 4, 128, 0, stream>>>(xh, wqh, qh, kh, vTh);

  // Attention: B*H*(N/16) = 3072 blocks
  attn_kernel<<<kB * kH * (kN / 16), 128, 0, stream>>>(
      qh, kh, vTh, prev, bp_w1, bp_b1, bp_w2, bp_b2, cur, aoh);

  // Projection: (4096/16)*(768/64) = 3072 strips, 4 waves/block
  proj_gemm_kernel<<<3072 / 4, 128, 0, stream>>>(aoh, wph, proj_b, out);
}
